// GAT_1614907703894
// MI455X (gfx1250) — matrix operand order repaired and checked by reading, not compiled, not run on hardware
//
#include <hip/hip_runtime.h>
#include <hip/hip_bf16.h>
#include <math.h>

// ---------------------------------------------------------------------------
// Types for CDNA5 WMMA (wave32, 16x16x32 bf16 -> f32)
// ---------------------------------------------------------------------------
typedef __attribute__((ext_vector_type(16))) __bf16          v16bf;
typedef __attribute__((ext_vector_type(16))) unsigned short  v16us;
typedef __attribute__((ext_vector_type(8)))  unsigned short  v8us;   // one b128
typedef __attribute__((ext_vector_type(8)))  float           v8f;

#define GAT_N      50000
#define GAT_E      800000
#define GAT_ETEST  100000
#define NEG_SLOPE  0.2f

// float -> bf16 raw bits, round-to-nearest-even
__device__ __forceinline__ unsigned short f2bf(float f) {
    unsigned u = __float_as_uint(f);
    u += 0x7FFFu + ((u >> 16) & 1u);
    return (unsigned short)(u >> 16);
}

// monotone order-preserving float <-> uint key (atomicMax-based segment max)
__device__ __forceinline__ unsigned fkey(float f) {
    unsigned u = __float_as_uint(f);
    return (u & 0x80000000u) ? ~u : (u | 0x80000000u);
}
__device__ __forceinline__ float fkey_dec(unsigned k) {
    unsigned u = (k & 0x80000000u) ? (k & 0x7FFFFFFFu) : ~k;
    return __uint_as_float(u);
}

// ---------------------------------------------------------------------------
// Pre-passes: f32 -> bf16 cast, and weight transpose+cast (W[K,Nc] -> Wt[Nc,K])
// ---------------------------------------------------------------------------
__global__ void k_cast_bf16(const float* __restrict__ in,
                            unsigned* __restrict__ out2, size_t npairs) {
    size_t i = (size_t)blockIdx.x * blockDim.x + threadIdx.x;
    if (i >= npairs) return;
    float2 v = ((const float2*)in)[i];
    out2[i] = (unsigned)f2bf(v.x) | ((unsigned)f2bf(v.y) << 16);
}

__global__ void k_transpose_cast(const float* __restrict__ W,
                                 unsigned short* __restrict__ Wt,
                                 int K, int Nc) {
    int idx = blockIdx.x * blockDim.x + threadIdx.x;
    if (idx >= K * Nc) return;
    int k = idx / Nc, n = idx - k * Nc;
    Wt[(size_t)n * K + k] = f2bf(W[idx]);
}

// ---------------------------------------------------------------------------
// GEMM: C[M,Nc] = A[M,K] @ B[K,Nc]; A as bf16 row-major, B pre-transposed bf16
// Bt[Nc,K].  One wave per 16xNc output strip (Nc == 64 -> 4 WMMA tiles),
// 4 waves per block.  M%16==0, K%32==0.  Fragment layouts per ISA 7.12.2:
//   A lane: row = m0+rc; K pairs {half*8..+8} and {16+half*8..+8}  -> 2x b128
//   B lane: col = n0+rc; K runs {half*8..+8} and {16+half*8..+8} (Bt row) -> 2x b128
// ---------------------------------------------------------------------------
__global__ void __launch_bounds__(128)
k_gemm_bf16(const unsigned short* __restrict__ A,
            const unsigned short* __restrict__ Bt,
            float* __restrict__ C, int M, int K, int Nc) {
    const int lane = threadIdx.x & 31;
    const int m0   = (blockIdx.x * 4 + threadIdx.y) * 16;
    if (m0 >= M) return;                      // uniform across the wave
    const int half = lane >> 4;               // 0: lanes 0-15, 1: lanes 16-31
    const int rc   = lane & 15;               // A row / B,C column within tile

    v8f acc0 = {}, acc1 = {}, acc2 = {}, acc3 = {};
    const unsigned short* Ar = A + (size_t)(m0 + rc) * K;

    for (int kk = 0; kk < K; kk += 32) {
        // A fragment: two contiguous 16B chunks
        v8us alo = *(const v8us*)(Ar + kk + half * 8);
        v8us ahi = *(const v8us*)(Ar + kk + 16 + half * 8);
        v16bf a = __builtin_bit_cast(v16bf,
            __builtin_shufflevector(alo, ahi, 0,1,2,3,4,5,6,7,
                                              8,9,10,11,12,13,14,15));
        #pragma unroll
        for (int nt = 0; nt < 4; ++nt) {
            const unsigned short* Br =
                Bt + (size_t)(nt * 16 + rc) * K + kk;
            v8us blo = *(const v8us*)(Br + half * 8);
            v8us bhi = *(const v8us*)(Br + 16 + half * 8);
            v16bf b = __builtin_bit_cast(v16bf,
                __builtin_shufflevector(blo, bhi, 0,1,2,3,4,5,6,7,
                                                  8,9,10,11,12,13,14,15));
            v8f cin = (nt == 0) ? acc0 : (nt == 1) ? acc1 : (nt == 2) ? acc2 : acc3;
            // (neg_a, A, neg_b, B, c_mod, C, reuse_a, reuse_b)
            v8f cout = __builtin_amdgcn_wmma_f32_16x16x32_bf16(
                           false, a, false, b, (short)0, cin, false, false);
            if      (nt == 0) acc0 = cout;
            else if (nt == 1) acc1 = cout;
            else if (nt == 2) acc2 = cout;
            else              acc3 = cout;
        }
    }
    // C fragment: lane rc = column, VGPR i -> row i (+8 for upper half-wave)
    #pragma unroll
    for (int i = 0; i < 8; ++i) {
        size_t row = (size_t)(m0 + i + half * 8);
        C[row * Nc +  0 + rc] = acc0[i];
        C[row * Nc + 16 + rc] = acc1[i];
        C[row * Nc + 32 + rc] = acc2[i];
        C[row * Nc + 48 + rc] = acc3[i];
    }
}

// a_src[n,h] = sum_c h[n,h,c]*att_src[h,c];  a_dst likewise
__global__ void k_att(const float* __restrict__ Hbuf,
                      const float* __restrict__ atts,
                      const float* __restrict__ attd,
                      float* __restrict__ asrc, float* __restrict__ adst,
                      int n, int Hh, int Cc) {
    int idx = blockIdx.x * blockDim.x + threadIdx.x;
    if (idx >= n * Hh) return;
    int node = idx / Hh, h = idx - node * Hh;
    const float* hp = Hbuf + (size_t)node * Hh * Cc + h * Cc;
    float s = 0.f, d = 0.f;
    for (int c = 0; c < Cc; ++c) {
        float v = hp[c];
        s = fmaf(v, atts[h * Cc + c], s);
        d = fmaf(v, attd[h * Cc + c], d);
    }
    asrc[idx] = s;
    adst[idx] = d;
}

__global__ void k_zero(unsigned* __restrict__ p, size_t n) {
    size_t i = (size_t)blockIdx.x * blockDim.x + threadIdx.x;
    if (i < n) p[i] = 0u;
}

// edge enumeration helper: t in [0, E+N); t >= E are self-loops (i,i)
__device__ __forceinline__ void edge_sd(const int* __restrict__ ei,
                                        long long t, int E_, int& s, int& d) {
    if (t < E_) { s = ei[t]; d = ei[(size_t)E_ + t]; }
    else        { s = d = (int)(t - E_); }
}

__device__ __forceinline__ float leaky(float a) {
    return a < 0.f ? NEG_SLOPE * a : a;
}

// pass 1: per-dst segment max of leaky_relu(a_src[s]+a_dst[d])
__global__ void k_edge_max(const float* __restrict__ asrc,
                           const float* __restrict__ adst,
                           const int* __restrict__ ei,
                           int E_, int n, int Hh, unsigned* __restrict__ mkey) {
    long long idx = (long long)blockIdx.x * blockDim.x + threadIdx.x;
    long long tot = (long long)(E_ + n) * Hh;
    if (idx >= tot) return;
    int h = (int)(idx % Hh);
    long long t = idx / Hh;
    int s, d; edge_sd(ei, t, E_, s, d);
    float a = leaky(asrc[(size_t)s * Hh + h] + adst[(size_t)d * Hh + h]);
    atomicMax(&mkey[(size_t)d * Hh + h], fkey(a));
}

// pass 2: denom[d,h] += exp(alpha - m[d,h])
__global__ void k_edge_denom(const float* __restrict__ asrc,
                             const float* __restrict__ adst,
                             const int* __restrict__ ei,
                             const unsigned* __restrict__ mkey,
                             float* __restrict__ denom,
                             int E_, int n, int Hh) {
    long long idx = (long long)blockIdx.x * blockDim.x + threadIdx.x;
    long long tot = (long long)(E_ + n) * Hh;
    if (idx >= tot) return;
    int h = (int)(idx % Hh);
    long long t = idx / Hh;
    int s, d; edge_sd(ei, t, E_, s, d);
    float a = leaky(asrc[(size_t)s * Hh + h] + adst[(size_t)d * Hh + h]);
    float m = fkey_dec(mkey[(size_t)d * Hh + h]);
    atomicAdd(&denom[(size_t)d * Hh + h], __expf(a - m));
}

// pass 3: out[d,h,:] += h[s,h,:] * exp(alpha-m)/denom
__global__ void k_edge_msg(const float* __restrict__ asrc,
                           const float* __restrict__ adst,
                           const int* __restrict__ ei,
                           const unsigned* __restrict__ mkey,
                           const float* __restrict__ denom,
                           const float* __restrict__ Hsrc,
                           float* __restrict__ out,
                           int E_, int n, int Hh, int Cc) {
    long long idx = (long long)blockIdx.x * blockDim.x + threadIdx.x;
    long long tot = (long long)(E_ + n) * Hh;
    if (idx >= tot) return;
    int h = (int)(idx % Hh);
    long long t = idx / Hh;
    int s, d; edge_sd(ei, t, E_, s, d);
    float a = leaky(asrc[(size_t)s * Hh + h] + adst[(size_t)d * Hh + h]);
    float m = fkey_dec(mkey[(size_t)d * Hh + h]);
    float e = __expf(a - m);
    float coef = e / fmaxf(denom[(size_t)d * Hh + h], 1e-16f);
    const float* hs = Hsrc + ((size_t)s * Hh + h) * Cc;
    float*       op = out  + ((size_t)d * Hh + h) * Cc;
    for (int c = 0; c < Cc; ++c)
        atomicAdd(&op[c], hs[c] * coef);
}

// z = (in + bias), optional ELU (alpha=1); optional fused bf16 copy of z
__global__ void k_bias_act(const float* __restrict__ in,
                           const float* __restrict__ bias,
                           float* __restrict__ outz,
                           unsigned short* __restrict__ outbf, // may be null
                           int n, int F, int do_elu) {
    size_t idx = (size_t)blockIdx.x * blockDim.x + threadIdx.x;
    if (idx >= (size_t)n * F) return;
    int f = (int)(idx % F);
    float v = in[idx] + bias[f];
    if (do_elu) v = v > 0.f ? v : (__expf(v) - 1.0f);
    outz[idx] = v;
    if (outbf) outbf[idx] = f2bf(v);
}

// logits[t] = dot64(z[src], z[dst]) over concat(pos, neg) test edges
__global__ void k_logits(const float* __restrict__ z,
                         const int* __restrict__ pos,
                         const int* __restrict__ neg,
                         float* __restrict__ out, int Et) {
    int t = blockIdx.x * blockDim.x + threadIdx.x;
    if (t >= 2 * Et) return;
    int s, d;
    if (t < Et) { s = pos[t]; d = pos[Et + t]; }
    else        { int u = t - Et; s = neg[u]; d = neg[Et + u]; }
    const float4* zs = (const float4*)(z + (size_t)s * 64);
    const float4* zd = (const float4*)(z + (size_t)d * 64);
    float acc = 0.f;
    #pragma unroll
    for (int i = 0; i < 16; ++i) {
        float4 a = zs[i], b = zd[i];
        acc += a.x * b.x + a.y * b.y + a.z * b.z + a.w * b.w;
    }
    out[t] = acc;
}

// ---------------------------------------------------------------------------
extern "C" void kernel_launch(void* const* d_in, const int* in_sizes, int n_in,
                              void* d_out, int out_size, void* d_ws, size_t ws_size,
                              hipStream_t stream) {
    (void)in_sizes; (void)n_in; (void)out_size; (void)ws_size;
    const float* x     = (const float*)d_in[0];   // [N,128]
    const int*   ei    = (const int*)  d_in[1];   // [2,E]
    const int*   pos   = (const int*)  d_in[2];   // [2,E_TEST]
    const int*   neg   = (const int*)  d_in[3];   // [2,E_TEST]
    const float* W1    = (const float*)d_in[4];   // [128,64]
    const float* as1w  = (const float*)d_in[5];   // [8,8]
    const float* ad1w  = (const float*)d_in[6];   // [8,8]
    const float* b1    = (const float*)d_in[7];   // [64]
    const float* W2    = (const float*)d_in[8];   // [64,64]
    const float* as2w  = (const float*)d_in[9];   // [1,64]
    const float* ad2w  = (const float*)d_in[10];  // [1,64]
    const float* b2    = (const float*)d_in[11];  // [64]
    float* logits = (float*)d_out;                // [200000]

    const int N = GAT_N, E = GAT_E, ET = GAT_ETEST;

    // ---- workspace layout: f32 region, then bf16 region (~78 MB) ----
    float*    h1  = (float*)d_ws;                    // N*64
    float*    z1  = h1  + (size_t)N * 64;            // N*64 (accum -> elu)
    float*    h2  = z1  + (size_t)N * 64;            // N*64
    float*    z2  = h2  + (size_t)N * 64;            // N*64 (accum -> +bias)
    float*    as1 = z2  + (size_t)N * 64;            // N*8
    float*    ad1 = as1 + (size_t)N * 8;             // N*8
    unsigned* mk1 = (unsigned*)(ad1 + (size_t)N * 8);// N*8
    float*    dn1 = (float*)(mk1 + (size_t)N * 8);   // N*8
    float*    as2 = dn1 + (size_t)N * 8;             // N
    float*    ad2 = as2 + N;                         // N
    unsigned* mk2 = (unsigned*)(ad2 + N);            // N
    float*    dn2 = (float*)(mk2 + N);               // N
    unsigned short* xbf  = (unsigned short*)(dn2 + N);   // N*128
    unsigned short* z1bf = xbf  + (size_t)N * 128;       // N*64
    unsigned short* w1t  = z1bf + (size_t)N * 64;        // 64*128 (W1^T)
    unsigned short* w2t  = w1t  + (size_t)64 * 128;      // 64*64  (W2^T)

    const int TB = 256;
    const long long EL1 = (long long)(E + N) * 8;    // edge*head work, layer 1
    const long long EL2 = (long long)(E + N);        // layer 2 (H=1)
    dim3 gemm_blk(32, 4);                            // 4 waves per block
    const int gemm_grid = (N / 16 + 3) / 4;

    // ---- operand preparation (bf16 casts + weight transposes) ----
    k_cast_bf16<<<((size_t)N * 64 + TB - 1) / TB, TB, 0, stream>>>(
        x, (unsigned*)xbf, (size_t)N * 64);          // N*128 floats = N*64 pairs
    k_transpose_cast<<<(128 * 64 + TB - 1) / TB, TB, 0, stream>>>(W1, w1t, 128, 64);
    k_transpose_cast<<<(64 * 64 + TB - 1) / TB, TB, 0, stream>>>(W2, w2t, 64, 64);

    // ---- layer 1 ----
    k_gemm_bf16<<<gemm_grid, gemm_blk, 0, stream>>>(xbf, w1t, h1, N, 128, 64);
    k_att<<<(N * 8 + TB - 1) / TB, TB, 0, stream>>>(h1, as1w, ad1w, as1, ad1, N, 8, 8);
    k_zero<<<((size_t)N * 64 + TB - 1) / TB, TB, 0, stream>>>((unsigned*)z1, (size_t)N * 64);
    k_zero<<<((size_t)N * 16 + TB - 1) / TB, TB, 0, stream>>>(mk1, (size_t)N * 16); // mk1+dn1
    k_edge_max  <<<(EL1 + TB - 1) / TB, TB, 0, stream>>>(as1, ad1, ei, E, N, 8, mk1);
    k_edge_denom<<<(EL1 + TB - 1) / TB, TB, 0, stream>>>(as1, ad1, ei, mk1, dn1, E, N, 8);
    k_edge_msg  <<<(EL1 + TB - 1) / TB, TB, 0, stream>>>(as1, ad1, ei, mk1, dn1, h1, z1, E, N, 8, 8);
    k_bias_act<<<((size_t)N * 64 + TB - 1) / TB, TB, 0, stream>>>(z1, b1, z1, z1bf, N, 64, 1);

    // ---- layer 2 ----
    k_gemm_bf16<<<gemm_grid, gemm_blk, 0, stream>>>(z1bf, w2t, h2, N, 64, 64);
    k_att<<<(N + TB - 1) / TB, TB, 0, stream>>>(h2, as2w, ad2w, as2, ad2, N, 1, 64);
    k_zero<<<((size_t)N * 64 + TB - 1) / TB, TB, 0, stream>>>((unsigned*)z2, (size_t)N * 64);
    k_zero<<<((size_t)N * 2 + TB - 1) / TB, TB, 0, stream>>>(mk2, (size_t)N * 2);   // mk2+dn2
    k_edge_max  <<<(EL2 + TB - 1) / TB, TB, 0, stream>>>(as2, ad2, ei, E, N, 1, mk2);
    k_edge_denom<<<(EL2 + TB - 1) / TB, TB, 0, stream>>>(as2, ad2, ei, mk2, dn2, E, N, 1);
    k_edge_msg  <<<(EL2 + TB - 1) / TB, TB, 0, stream>>>(as2, ad2, ei, mk2, dn2, h2, z2, E, N, 1, 64);
    k_bias_act<<<((size_t)N * 64 + TB - 1) / TB, TB, 0, stream>>>(z2, b2, z2, (unsigned short*)nullptr, N, 64, 0);

    // ---- link-prediction logits ----
    k_logits<<<(2 * ET + TB - 1) / TB, TB, 0, stream>>>(z2, pos, neg, logits, ET);
}
